// HSCRF_2559800508781
// MI455X (gfx1250) — matrix operand order repaired and checked by reading, not compiled, not run on hardware
//
#include <hip/hip_runtime.h>
#include <hip/hip_bf16.h>
#include <cstdint>

// ---------------------------------------------------------------------------
// Semi-Markov CRF forward (log-partition), B=64, L=200, T=6, K=6.
//
// Scaled-probability reformulation: carry p[j]=exp(alpha[e-k][y'] - m),
// j = k*6+y' (36 values, kept <= 1, running log-scale m). Precompute
// Sexp = exp(scores band) once (invalid span -> 0.0). Each step is then a
// pure matvec s[y] = sum_j p[j]*Sexp[j,y] -> v_wmma_f32_16x16x32_f16
// (two chained K=32 WMMAs, row 0 of A = p), followed by
// r = max(max_y s[y], 1);  p' = [s/r, shift(p)/r];  m += log(r).
// alpha_{e+1}[STOP] = m_new + log(p'[STOP]) (captured, logged once at end).
//
// p lives in LDS as f16 in A-fragment order (pa[0..35], pa[36..79] == 0), so
// A fragments are plain per-lane ds_load_b128s. The whole loop body is
// EXEC-uniform: WMMA guarantees c0 == 0 on lanes 6..31, so the s-staging
// store is unconditional, and the shift store re-writes the zero pad.
//
// Pass 1 writes Sexp in this code's B-fragment per-lane layout, contiguous k order as in A
// (lane l: N = l%16, half index i <-> K = i + 16*(l>=16)); pass 2 streams it
// global->LDS with CDNA5 async copies (global_load_async_to_lds_b128),
// triple-buffered with prefetch distance 2 and partial s_wait_asynccnt 4
// (async copies complete in order per wave).
//
// Workspace:  [0,256)    : 64 float per-batch results
//             [256, ...) : Sexp band, B*L*2*32*16 halfs  (~26.2 MB)
// ---------------------------------------------------------------------------

typedef __attribute__((ext_vector_type(16))) _Float16 v16h;
typedef __attribute__((ext_vector_type(8)))  _Float16 v8h;
typedef __attribute__((ext_vector_type(8)))  float    v8f;

#define Bn 64
#define Ln 200
#define Tn 6
#define START_ID 4
#define STOP_ID  5
#define W_OFF 256            // byte offset of Sexp band inside d_ws
#define STEP_BYTES 2048      // 2 chunks * 32 lanes * 16 halfs * 2B per (b,e)

__device__ __forceinline__ uint32_t lds_addr32(const void* p) {
  // shared->generic cast puts the LDS byte offset in the low 32 bits
  return (uint32_t)(uintptr_t)p;
}

__device__ __forceinline__ void async_b128(uint32_t lds, uint64_t gaddr) {
  asm volatile("global_load_async_to_lds_b128 %0, %1, off"
               :: "v"(lds), "v"(gaddr) : "memory");
}

__device__ __forceinline__ void wait_async0() {
  asm volatile("s_wait_asynccnt 0" ::: "memory");
}
__device__ __forceinline__ void wait_async4() {
  asm volatile("s_wait_asynccnt 4" ::: "memory");
}

// ---------------------------------------------------------------------------
// Pass 1: Sexp[b,e,chunk,lane,h] = exp(scores[b, e-k, e, y', y]) in B layout.
//   N = lane%16 (y), half = lane/16, Klocal = h + 16*half, j = chunk*32+Klocal,
//   k = j/6, y' = j%6; zero if N>=6, j>=36, or e-k<0.
// ---------------------------------------------------------------------------
__global__ void hscrf_expand(const float* __restrict__ scores,
                             _Float16* __restrict__ W) {
  int tid   = blockIdx.x * blockDim.x + threadIdx.x;   // 819200 total
  int lane  = tid & 31;
  int chunk = (tid >> 5) & 1;
  int e     = (tid >> 6) % Ln;
  int b     = tid / (Ln * 64);
  int N     = lane & 15;
  int half  = lane >> 4;

  v16h v;
#pragma unroll
  for (int h = 0; h < 16; ++h) {
    int j = chunk * 32 + h + 16 * half;
    float val = 0.0f;
    if (N < Tn && j < 36) {
      int k  = j / 6;
      int yp = j - 6 * k;
      int s  = e - k;
      if (s >= 0) {
        size_t idx = (((size_t)b * Ln + s) * Ln + e) * (Tn * Tn) + yp * Tn + N;
        val = __expf(scores[idx]);
      }
    }
    v[h] = (_Float16)val;
  }
  size_t wslot = (((size_t)b * Ln + e) * 2 + chunk) * 32 + lane;
  *(v16h*)(W + wslot * 16) = v;
}

// ---------------------------------------------------------------------------
// Pass 2: one wave32 per batch; sequential scan with WMMA matvec and
// async-LDS triple buffering (prefetch distance 2) of the band.
// ---------------------------------------------------------------------------
__global__ void __launch_bounds__(32)
hscrf_scan(const _Float16* __restrict__ W, const int* __restrict__ mask,
           float* __restrict__ res) {
  __shared__ __align__(32) _Float16 sbuf[3][1024];  // 3 x 2048 B ring buffer
  __shared__ __align__(16) _Float16 pa[80];         // p as f16; [36..79] == 0
  __shared__ __align__(16) float    st[32];         // s staging (c0 all lanes)

  const int b     = blockIdx.x;
  const int lane  = threadIdx.x;
  const int maskb = mask[b];

  pa[lane]      = (_Float16)((lane == START_ID) ? 1.0f : 0.0f);
  pa[32 + lane] = (_Float16)0.0f;                   // covers 32..63
  if (lane < 16) pa[64 + lane] = (_Float16)0.0f;    // covers 64..79

  // constant per-lane A-fragment addresses (half indices into pa)
  const int a0lo = (lane == 0) ? 0 : ((lane == 16) ? 16 : 48);
  const int a0hi = a0lo + 8;                        // 8/24 data, 56 zeros
  const int a1lo = (lane == 0) ? 32 : 48;           // pa[36..39] are zero pad
  const int shq  = (lane < 4) ? (26 + lane) : 40;   // shift src; 40 is zero

  float m = 0.0f, mcap = 0.0f, pcap = 1.0f;
  const uint64_t gbase = (uint64_t)(uintptr_t)W + (uint64_t)b * Ln * STEP_BYTES;
  const uint32_t lo = lane * 32;

#define PREFETCH(buf_idx, step)                                              \
  do {                                                                       \
    uint32_t lb = lds_addr32(&sbuf[(buf_idx)][0]);                           \
    uint64_t gb = gbase + (uint64_t)(step) * STEP_BYTES;                     \
    async_b128(lb + lo,             gb + lo);                                \
    async_b128(lb + lo + 16,        gb + lo + 16);                           \
    async_b128(lb + 1024 + lo,      gb + 1024 + lo);                         \
    async_b128(lb + 1024 + lo + 16, gb + 1024 + lo + 16);                    \
  } while (0)

  PREFETCH(0, 0);                        // distance-2 pipeline prologue
  PREFETCH(1, 1);

  int cur = 0;
  for (int e = 0; e < Ln; ++e) {
    // oldest 4 async copies (step e) complete; step e+1's stay in flight
    if (e + 1 < Ln) wait_async4(); else wait_async0();

    int nxt = cur + 1; if (nxt == 3) nxt = 0;
    int pfb = nxt + 1; if (pfb == 3) pfb = 0;
    if (e + 2 < Ln) PREFETCH(pfb, e + 2);  // uniform branch

    // B fragments: per-lane 32B, already in WMMA layout
    v16h bf0 = *(const v16h*)&sbuf[cur][lane * 16];
    v16h bf1 = *(const v16h*)&sbuf[cur][512 + lane * 16];

    // A fragments: straight b128 LDS loads, zero lanes read the zero region
    v8h alo = *(const v8h*)&pa[a0lo];
    v8h ahi = *(const v8h*)&pa[a0hi];
    v8h blo = *(const v8h*)&pa[a1lo];
    v8h zz  = *(const v8h*)&pa[64];
    v16h a0 = __builtin_shufflevector(alo, ahi, 0,1,2,3,4,5,6,7,
                                                8,9,10,11,12,13,14,15);
    v16h a1 = __builtin_shufflevector(blo, zz,  0,1,2,3,4,5,6,7,
                                                8,9,10,11,12,13,14,15);

    v8f c = {0.f, 0.f, 0.f, 0.f, 0.f, 0.f, 0.f, 0.f};
    c = __builtin_amdgcn_wmma_f32_16x16x32_f16(false, a0, false, bf0,
                                               (short)0, c, false, false);
    c = __builtin_amdgcn_wmma_f32_16x16x32_f16(false, a1, false, bf1,
                                               (short)0, c, false, false);
    const float c0 = c[0];       // s[y] on lane y; exactly 0 on lanes 6..31

    st[lane] = c0;               // uniform store; st[6..7] == 0 by WMMA math

    // old p entries (read before any pa store; slots 36/40 are hard zeros)
    const float oldp = (float)pa[lane >= 6 ? lane - 6 : 36];
    const float oldq = (float)pa[shq];

    // r = max(max_y s[y], 1): keeps all entries <= 1
    float4 s0 = *(const float4*)&st[0];
    float4 s1 = *(const float4*)&st[4];
    float rs = fmaxf(fmaxf(fmaxf(s0.x, s0.y), fmaxf(s0.z, s0.w)),
                     fmaxf(fmaxf(s1.x, s1.y), fmaxf(s1.z, s1.w)));
    const float r   = fmaxf(rs, 1.0f);
    const float inv = __builtin_amdgcn_rcpf(r);   // 1-ulp v_rcp_f32

    // shift + rescale, both stores wave-uniform (lanes >= 4 rewrite zeros)
    const float nv = ((lane < Tn) ? c0 : oldp) * inv;
    pa[lane]      = (_Float16)nv;
    pa[32 + lane] = (_Float16)(oldq * inv);

    m += __logf(r);
    const bool hit = (lane == STOP_ID) & ((e + 1) == maskb);
    pcap = hit ? nv : pcap;              // p'[STOP] at the masked position
    mcap = hit ? m  : mcap;              // matching scale
    cur = nxt;
  }
#undef PREFETCH

  if (lane == STOP_ID) res[b] = mcap + __logf(pcap);
}

// ---------------------------------------------------------------------------
// Pass 3: deterministic serial reduction (avoids float-atomic nondeterminism).
// ---------------------------------------------------------------------------
__global__ void hscrf_reduce(const float* __restrict__ res,
                             float* __restrict__ out) {
  if (threadIdx.x == 0 && blockIdx.x == 0) {
    float s = 0.0f;
    for (int i = 0; i < Bn; ++i) s += res[i];
    out[0] = s;
  }
}

extern "C" void kernel_launch(void* const* d_in, const int* in_sizes, int n_in,
                              void* d_out, int out_size, void* d_ws, size_t ws_size,
                              hipStream_t stream) {
  const float* scores = (const float*)d_in[0];
  const int*   mask   = (const int*)d_in[1];   // JAX w/o x64: int64 -> int32
  float*       out    = (float*)d_out;
  float*       res    = (float*)d_ws;
  _Float16*    W      = (_Float16*)((char*)d_ws + W_OFF);
  // ws requirement: 256 + 64*200*2048 bytes ~= 26.2 MB

  const int total = Bn * Ln * 2 * 32;          // 819200
  hscrf_expand<<<total / 256, 256, 0, stream>>>(scores, W);
  hscrf_scan  <<<Bn, 32, 0, stream>>>(W, mask, res);
  hscrf_reduce<<<1, 1, 0, stream>>>(res, out);
}
